// JointModel_61143154425964
// MI455X (gfx1250) — matrix operand order repaired and checked by reading, not compiled, not run on hardware
//
#include <hip/hip_runtime.h>
#include <hip/hip_bf16.h>

// ---------------------------------------------------------------------------
// Types
// ---------------------------------------------------------------------------
typedef __attribute__((ext_vector_type(16))) __bf16        v16bf;
typedef __attribute__((ext_vector_type(8)))  float         v8f;
typedef __attribute__((ext_vector_type(8)))  unsigned int  v8u;
typedef __attribute__((ext_vector_type(4)))  unsigned int  u32x4;
typedef __attribute__((ext_vector_type(4)))  int           i32x4;
typedef __attribute__((ext_vector_type(8)))  int           i32x8;

#define NH_      16
#define NKV_     4
#define HD_      128
#define B_       4
#define SQ_VLM_  1024
#define SQ_ACT_  64
#define S_CACHE_ 256
#define S_TOTAL_ 1344          // 256 + 1024 + 64
#define SQ_ALL_  1088          // 1024 + 64 (mask rows)

#define KPAD_    136           // LDS row stride (shorts) for 128-wide K tiles
#define VPAD_    72            // LDS row stride (shorts) for 64-wide V^T tiles

__device__ __forceinline__ unsigned short f2bf(float x) {
    unsigned int u = __builtin_bit_cast(unsigned int, x);
    unsigned int r = u + 0x7FFFu + ((u >> 16) & 1u);
    return (unsigned short)(r >> 16);
}
__device__ __forceinline__ float bf2f(unsigned short s) {
    unsigned int u = ((unsigned int)s) << 16;
    return __builtin_bit_cast(float, u);
}
__device__ __forceinline__ void store_out(float* C, size_t i, float v)          { C[i] = v; }
__device__ __forceinline__ void store_out(unsigned short* C, size_t i, float v) { C[i] = f2bf(v); }

// A fragment (16x32 bf16, MxK): lane = row (m=lane&15); VGPR v holds K pair
__device__ __forceinline__ v16bf load_a_frag(const unsigned short* base, int ldr, int kOff) {
    const int lane = threadIdx.x & 31, m = lane & 15, half = lane >> 4;
    v8u t;
#pragma unroll
    for (int v = 0; v < 8; ++v) {
        const int k0 = ((v < 4) ? (2 * v) : (8 + 2 * v)) + 8 * half + kOff;
        t[v] = *(const unsigned int*)(base + (size_t)m * ldr + k0);
    }
    return __builtin_bit_cast(v16bf, t);
}
// B fragment (32x16 bf16, KxN) from an N-major tile base[n][k] (pairs contiguous)
__device__ __forceinline__ v16bf load_b_frag(const unsigned short* base, int ldr, int kOff) {
    const int lane = threadIdx.x & 31, n = lane & 15, half = lane >> 4;
    v8u t;
#pragma unroll
    for (int v = 0; v < 8; ++v) {
        const int k0 = ((v < 4) ? (2 * v) : (8 + 2 * v)) + 8 * half + kOff;
        t[v] = *(const unsigned int*)(base + (size_t)n * ldr + k0);
    }
    return __builtin_bit_cast(v16bf, t);
}

// 16-lane reductions via ds_swizzle (xor mode, and_mask=0x1f)
template <int M>
__device__ __forceinline__ float swz_xor(float x) {
    const int y = __builtin_amdgcn_ds_swizzle(__builtin_bit_cast(int, x), (M << 10) | 0x1f);
    return __builtin_bit_cast(float, y);
}
__device__ __forceinline__ float red16max(float x) {
    x = fmaxf(x, swz_xor<1>(x)); x = fmaxf(x, swz_xor<2>(x));
    x = fmaxf(x, swz_xor<4>(x)); x = fmaxf(x, swz_xor<8>(x));
    return x;
}
__device__ __forceinline__ float red16sum(float x) {
    x += swz_xor<1>(x); x += swz_xor<2>(x);
    x += swz_xor<4>(x); x += swz_xor<8>(x);
    return x;
}

// ---------------------------------------------------------------------------
// TDM: async-load a 2D bf16 tile into LDS with row padding.
//   tile0 = fastest-dim tile length (elems), tile1 = #rows
//   tdim0/tdim1 = tensor dims (OOB), stride0 = row stride (elems)
//   padIntv/padAmt = D# pad codes (interval 2<<i DWORDs, amount amt+1 DWORDs)
// ---------------------------------------------------------------------------
__device__ __forceinline__ void tdm_load_2d(unsigned ldsOff, const unsigned short* gptr,
                                            unsigned tile0, unsigned tile1,
                                            unsigned tdim0, unsigned tdim1, unsigned stride0,
                                            unsigned padIntv, unsigned padAmt) {
    const unsigned long long ga = (unsigned long long)(size_t)gptr;
    u32x4 g0;
    g0[0] = 1u;                                                  // count=1, user mode
    g0[1] = ldsOff;                                              // lds_addr (bytes)
    g0[2] = (unsigned)(ga & 0xFFFFFFFFu);                        // global_addr[31:0]
    g0[3] = (unsigned)((ga >> 32) & 0x01FFFFFFu) | (2u << 30);   // addr[56:32] | type=2
    i32x8 g1;
    g1[0] = (int)((1u << 16) | (1u << 20) | (padIntv << 22) | (padAmt << 25)); // 2B, pad
    g1[1] = (int)((tdim0 & 0xFFFFu) << 16);                      // tensor_dim0 lo16
    g1[2] = (int)(((tdim0 >> 16) & 0xFFFFu) | ((tdim1 & 0xFFFFu) << 16));
    g1[3] = (int)(((tdim1 >> 16) & 0xFFFFu) | ((tile0 & 0xFFFFu) << 16));
    g1[4] = (int)(tile1 & 0xFFFFu);                              // tile_dim1, tile_dim2=0
    g1[5] = (int)stride0;                                        // tensor_dim0_stride lo32
    g1[6] = 0;
    g1[7] = 0;
    const i32x4 z4 = {0, 0, 0, 0};
#if defined(__clang_major__) && (__clang_major__ >= 23)
    const i32x8 z8 = {0, 0, 0, 0, 0, 0, 0, 0};
    __builtin_amdgcn_tensor_load_to_lds(g0, g1, z4, z4, z8, 0);
#else
    __builtin_amdgcn_tensor_load_to_lds(g0, g1, z4, z4, 0);
#endif
}

// ---------------------------------------------------------------------------
// Kernel 1: f32 KV cache -> bf16. K row-major [B][NKV][S_TOTAL][HD],
//           V transposed [B][NKV][HD][S_TOTAL].
// ---------------------------------------------------------------------------
__global__ void cvt_cache_kernel(const float* __restrict__ kc, const float* __restrict__ vc,
                                 unsigned short* __restrict__ Kb, unsigned short* __restrict__ Vb) {
    const int idx = blockIdx.x * blockDim.x + threadIdx.x;
    if (idx >= B_ * NKV_ * S_CACHE_ * HD_) return;
    const int d  = idx & (HD_ - 1);
    const int s  = (idx >> 7) & (S_CACHE_ - 1);
    const int bh = idx >> 15;
    Kb[((size_t)bh * S_TOTAL_ + s) * HD_ + d]  = f2bf(kc[idx]);
    Vb[((size_t)bh * HD_ + d) * S_TOTAL_ + s]  = f2bf(vc[idx]);
}

// ---------------------------------------------------------------------------
// Kernel 2: bf16 WMMA GEMM  C[M,N] = A[M,K] @ B[K,N]; 64x64 tile, BK=32.
// mode 0: row-major ; mode 1: K scatter [B][numH][S_TOTAL][128]+sOff ;
// mode 2: Q scatter [B][NH][64][128] ; mode 3: V^T scatter [B][numH][128][S_TOTAL]
// ---------------------------------------------------------------------------
template <typename AT, typename OT>
__global__ __launch_bounds__(128) void gemm_bf16(const AT* __restrict__ A,
                                                 const float* __restrict__ Bw,
                                                 OT* __restrict__ C,
                                                 int M, int N, int K,
                                                 int mode, int sOff, int seqLen, int numH) {
    __shared__ unsigned short As[64][40];    // [m][k], 80B rows (b64-aligned stores)
    __shared__ unsigned short Bst[64][40];   // [n][k], transposed

    const int tid  = threadIdx.x;
    const int wave = tid >> 5, lane = tid & 31, half = lane >> 4, ln = lane & 15;
    const int rowBase = blockIdx.x * 64;
    const int colBase = blockIdx.y * 64;

    v8f acc[4];
#pragma unroll
    for (int t = 0; t < 4; ++t) acc[t] = v8f{0.f,0.f,0.f,0.f,0.f,0.f,0.f,0.f};

    for (int kt = 0; kt < K; kt += 32) {
        __syncthreads();
#pragma unroll
        for (int i = 0; i < 4; ++i) {            // A tile 64x32
            const int flat = i * 128 + tid;
            const int m = flat >> 3, kq = flat & 7;
            unsigned int p0, p1;
            if constexpr (sizeof(AT) == 4) {
                const float4 a4 = *(const float4*)((const float*)A + (size_t)(rowBase + m) * K + kt + kq * 4);
                p0 = (unsigned)f2bf(a4.x) | ((unsigned)f2bf(a4.y) << 16);
                p1 = (unsigned)f2bf(a4.z) | ((unsigned)f2bf(a4.w) << 16);
            } else {
                const uint2 a2 = *(const uint2*)((const unsigned short*)A + (size_t)(rowBase + m) * K + kt + kq * 4);
                p0 = a2.x; p1 = a2.y;
            }
            *(uint2*)(&As[m][kq * 4]) = make_uint2(p0, p1);
        }
#pragma unroll
        for (int i = 0; i < 4; ++i) {            // B tile 32x64 -> transposed scatter
            const int flat = i * 128 + tid;
            const int k = flat >> 4, nq = flat & 15;
            const float4 b4 = *(const float4*)(Bw + (size_t)(kt + k) * N + colBase + nq * 4);
            Bst[nq * 4 + 0][k] = f2bf(b4.x);
            Bst[nq * 4 + 1][k] = f2bf(b4.y);
            Bst[nq * 4 + 2][k] = f2bf(b4.z);
            Bst[nq * 4 + 3][k] = f2bf(b4.w);
        }
        __syncthreads();

        const v16bf af = load_a_frag(&As[wave * 16][0], 40, 0);
#pragma unroll
        for (int t = 0; t < 4; ++t) {
            const v16bf bfr = load_b_frag(&Bst[16 * t][0], 40, 0);
            acc[t] = __builtin_amdgcn_wmma_f32_16x16x32_bf16(
                false, af, false, bfr, (short)0, acc[t], false, false);
        }
    }

#pragma unroll
    for (int t = 0; t < 4; ++t) {
        if (mode == 3) {
            // V^T: lane owns 8 consecutive seq rows at fixed output dim d
            const int col = colBase + 16 * t + ln;        // output channel
            const int hv = col >> 7, d = col & 127;
            const int row0 = rowBase + wave * 16 + 8 * half;  // multiple of 8
            const int b = row0 / seqLen, s0 = row0 % seqLen;
            unsigned int pk[4];
#pragma unroll
            for (int j = 0; j < 4; ++j)
                pk[j] = (unsigned)f2bf(acc[t][2 * j]) | ((unsigned)f2bf(acc[t][2 * j + 1]) << 16);
            unsigned short* dst = (unsigned short*)C +
                (((size_t)(b * numH + hv) * HD_ + d) * S_TOTAL_ + sOff + s0);
            *(uint4*)dst = make_uint4(pk[0], pk[1], pk[2], pk[3]);
        } else {
#pragma unroll
            for (int r = 0; r < 8; ++r) {
                const int row = rowBase + wave * 16 + r + 8 * half;
                const int col = colBase + 16 * t + ln;
                const float val = acc[t][r];
                if (mode == 0) {
                    store_out(C, (size_t)row * N + col, val);
                } else if (mode == 1) {
                    const int b = row / seqLen, s = row % seqLen;
                    const int hv = col >> 7, d = col & 127;
                    store_out(C, (((size_t)(b * numH + hv)) * S_TOTAL_ + sOff + s) * HD_ + d, val);
                } else {
                    const int b = row >> 6, s = row & 63;
                    const int hh = col >> 7, d = col & 127;
                    store_out(C, (((size_t)(b * NH_ + hh)) * SQ_ACT_ + s) * HD_ + d, val);
                }
            }
        }
    }
}

// ---------------------------------------------------------------------------
// Kernel 3: in-place RoPE on bf16 buffer [B][H][stot][128] (K and Q only)
// ---------------------------------------------------------------------------
__global__ void rope_kernel(unsigned short* __restrict__ buf, const int* __restrict__ pos,
                            int H, int s0, int len, int stot, int total) {
    const int idx = blockIdx.x * blockDim.x + threadIdx.x;
    if (idx >= total) return;
    const int d = idx & 63;
    const int s = (idx >> 6) % len;
    const int rest = (idx >> 6) / len;
    const int h = rest % H, b = rest / H;
    const float p = (float)pos[b * len + s];
    const float theta = p * __powf(10000.0f, -(float)d * (1.0f / 64.0f));
    const float cs = __cosf(theta), sn = __sinf(theta);
    unsigned short* ptr = buf + (((size_t)(b * H + h) * stot) + s0 + s) * HD_;
    const float x1 = bf2f(ptr[d]), x2 = bf2f(ptr[d + 64]);
    ptr[d]      = f2bf(x1 * cs - x2 * sn);
    ptr[d + 64] = f2bf(x2 * cs + x1 * sn);
}

// ---------------------------------------------------------------------------
// Kernel 4: flash attention, 1 block = (b, head), 4 waves x 16 query rows.
// K (row-major) and V^T (d-major) tiles staged by the TDM; both WMMA B-frags
// are contiguous-pair ds_load_b32. Consecutive WMMAs hit different
// accumulators (chunk-outer loops) to avoid RAW hazard NOPs.
// ---------------------------------------------------------------------------
__global__ __launch_bounds__(128) void attn_kernel(const unsigned short* __restrict__ Kbuf,
                                                   const unsigned short* __restrict__ Vbuf,
                                                   const unsigned short* __restrict__ Qbuf,
                                                   const float* __restrict__ mask,
                                                   unsigned short* __restrict__ Obuf) {
    __shared__ unsigned short Ks[64][KPAD_];    // [key][dim]
    __shared__ unsigned short Vt[128][VPAD_];   // [dim][key]
    __shared__ unsigned short Ps[4][16][72];    // per-wave P tile

    const int b  = blockIdx.x / NH_;
    const int h  = blockIdx.x % NH_;
    const int kv = h >> 2;
    const int tid = threadIdx.x;
    const int wave = tid >> 5, lane = tid & 31, half = lane >> 4, ln = lane & 15;

    const unsigned short* Qb = Qbuf + ((size_t)(b * NH_ + h) * SQ_ACT_ + wave * 16) * HD_;
    v16bf qf[4];
#pragma unroll
    for (int c = 0; c < 4; ++c) qf[c] = load_a_frag(Qb, HD_, 32 * c);

    v8f o[8];
#pragma unroll
    for (int nt = 0; nt < 8; ++nt) o[nt] = v8f{0.f,0.f,0.f,0.f,0.f,0.f,0.f,0.f};
    float Mr[8], Lr[8];
#pragma unroll
    for (int r = 0; r < 8; ++r) { Mr[r] = -1e30f; Lr[r] = 0.f; }

    const unsigned short* Kg = Kbuf + (size_t)(b * NKV_ + kv) * S_TOTAL_ * HD_;
    const unsigned short* Vg = Vbuf + (size_t)(b * NKV_ + kv) * HD_ * S_TOTAL_;
    const unsigned ldsK = (unsigned)(size_t)(&Ks[0][0]);
    const unsigned ldsV = (unsigned)(size_t)(&Vt[0][0]);
    const float* maskRow = mask + ((size_t)b * SQ_ALL_ + SQ_VLM_) * S_TOTAL_;

    for (int sb = 0; sb < S_TOTAL_ / 64; ++sb) {
        __syncthreads();                       // all waves done with Ks/Vt/Ps
        if (wave == 0) {
            // K: 64 keys x 128 dims, rows of 128 elems, pad 4DW per 64DW
            tdm_load_2d(ldsK, Kg + (size_t)sb * 64 * HD_, 128, 64, 128, S_TOTAL_, 128, 5, 3);
        } else if (wave == 1) {
            // V^T: 128 dims x 64 keys, rows of 64 elems from d-major tensor,
            // row stride S_TOTAL, pad 4DW per 32DW
            tdm_load_2d(ldsV, Vg + (size_t)sb * 64, 64, 128, S_TOTAL_, 128, S_TOTAL_, 4, 3);
        }
        if (wave < 2) __builtin_amdgcn_s_wait_tensorcnt(0);
        __syncthreads();                       // tiles visible to all waves

        // ---- scores (chunk-outer: consecutive WMMAs -> different accumulators)
        v8f sc[4];
#pragma unroll
        for (int t = 0; t < 4; ++t) sc[t] = v8f{0.f,0.f,0.f,0.f,0.f,0.f,0.f,0.f};
#pragma unroll
        for (int c = 0; c < 4; ++c) {
#pragma unroll
            for (int t = 0; t < 4; ++t) {
                const v16bf bfr = load_b_frag(&Ks[16 * t][0], KPAD_, 32 * c);
                sc[t] = __builtin_amdgcn_wmma_f32_16x16x32_bf16(
                    false, qf[c], false, bfr, (short)0, sc[t], false, false);
            }
        }

        // ---- scale, tanh soft-clamp, mask
#pragma unroll
        for (int t = 0; t < 4; ++t) {
#pragma unroll
            for (int r = 0; r < 8; ++r) {
                float s = sc[t][r] * 0.08838834764831845f;   // 1/sqrt(128)
                s = tanhf(s * 0.02f) * 50.0f;
                const int q   = wave * 16 + r + 8 * half;
                const int key = sb * 64 + 16 * t + ln;
                s += maskRow[(size_t)q * S_TOTAL_ + key];
                sc[t][r] = s;
            }
        }

        // ---- online softmax per query row
        float alpha[8];
#pragma unroll
        for (int r = 0; r < 8; ++r) {
            float mx = fmaxf(fmaxf(sc[0][r], sc[1][r]), fmaxf(sc[2][r], sc[3][r]));
            mx = red16max(mx);
            const float mnew = fmaxf(Mr[r], mx);
            alpha[r] = __expf(Mr[r] - mnew);
            Mr[r] = mnew;
            float rs = 0.f;
#pragma unroll
            for (int t = 0; t < 4; ++t) {
                const float p = __expf(sc[t][r] - mnew);
                sc[t][r] = p;
                rs += p;
            }
            rs = red16sum(rs);
            Lr[r] = Lr[r] * alpha[r] + rs;
        }
#pragma unroll
        for (int nt = 0; nt < 8; ++nt)
#pragma unroll
            for (int r = 0; r < 8; ++r) o[nt][r] *= alpha[r];

        // ---- P -> LDS (C layout -> row major), reload as A fragments
#pragma unroll
        for (int t = 0; t < 4; ++t)
#pragma unroll
            for (int r = 0; r < 8; ++r)
                Ps[wave][r + 8 * half][16 * t + ln] = f2bf(sc[t][r]);
        __syncthreads();

        v16bf pf[2];
#pragma unroll
        for (int c = 0; c < 2; ++c) pf[c] = load_a_frag(&Ps[wave][0][0], 72, 32 * c);

        // ---- O += P @ V  (V^T tile: contiguous-pair B-frags; chunk-outer)
#pragma unroll
        for (int c = 0; c < 2; ++c) {
#pragma unroll
            for (int nt = 0; nt < 8; ++nt) {
                const v16bf bfr = load_b_frag(&Vt[16 * nt][0], VPAD_, 32 * c);
                o[nt] = __builtin_amdgcn_wmma_f32_16x16x32_bf16(
                    false, pf[c], false, bfr, (short)0, o[nt], false, false);
            }
        }
    }

    // ---- normalize, write Obuf[b][q][h*128+d] (bf16)
#pragma unroll
    for (int nt = 0; nt < 8; ++nt) {
#pragma unroll
        for (int r = 0; r < 8; ++r) {
            const float val = o[nt][r] / Lr[r];
            const int q = wave * 16 + r + 8 * half;
            const int d = 16 * nt + ln;
            Obuf[((size_t)b * SQ_ACT_ + q) * (NH_ * HD_) + h * HD_ + d] = f2bf(val);
        }
    }
}

// ---------------------------------------------------------------------------
// Launch
// ---------------------------------------------------------------------------
extern "C" void kernel_launch(void* const* d_in, const int* in_sizes, int n_in,
                              void* d_out, int out_size, void* d_ws, size_t ws_size,
                              hipStream_t stream) {
    const float* mask    = (const float*)d_in[0];
    const int*   pos_vlm = (const int*)d_in[1];
    const int*   pos_act = (const int*)d_in[2];
    const float* h_vlm   = (const float*)d_in[3];
    const float* h_act   = (const float*)d_in[4];
    const float* k_cache = (const float*)d_in[5];
    const float* v_cache = (const float*)d_in[6];
    // d_in[7] = Wq_vlm : unused (only action-query rows reach the output)
    const float* Wk_vlm  = (const float*)d_in[8];
    const float* Wv_vlm  = (const float*)d_in[9];
    const float* Wq_act  = (const float*)d_in[10];
    const float* Wk_act  = (const float*)d_in[11];
    const float* Wv_act  = (const float*)d_in[12];
    const float* Wo_act  = (const float*)d_in[13];
    float* out = (float*)d_out;

    const size_t nKV = (size_t)B_ * NKV_ * S_TOTAL_ * HD_;
    const size_t nQ  = (size_t)B_ * NH_ * SQ_ACT_ * HD_;
    unsigned short* Kbuf = (unsigned short*)d_ws;   // [B][NKV][S_TOTAL][HD]
    unsigned short* Vbuf = Kbuf + nKV;              // [B][NKV][HD][S_TOTAL] (transposed)
    unsigned short* Qbuf = Vbuf + nKV;              // [B][NH][64][HD]
    unsigned short* Obuf = Qbuf + nQ;               // [B][64][NH*HD]

    const dim3 blk(128);

    cvt_cache_kernel<<<(B_ * NKV_ * S_CACHE_ * HD_ + 255) / 256, 256, 0, stream>>>(
        k_cache, v_cache, Kbuf, Vbuf);

    gemm_bf16<float, unsigned short><<<dim3(64, 8), blk, 0, stream>>>(h_vlm, Wk_vlm, Kbuf, 4096, 512, 2048, 1, S_CACHE_, SQ_VLM_, NKV_);
    gemm_bf16<float, unsigned short><<<dim3(64, 8), blk, 0, stream>>>(h_vlm, Wv_vlm, Vbuf, 4096, 512, 2048, 3, S_CACHE_, SQ_VLM_, NKV_);
    gemm_bf16<float, unsigned short><<<dim3(4, 8),  blk, 0, stream>>>(h_act, Wk_act, Kbuf, 256, 512, 2048, 1, S_CACHE_ + SQ_VLM_, SQ_ACT_, NKV_);
    gemm_bf16<float, unsigned short><<<dim3(4, 8),  blk, 0, stream>>>(h_act, Wv_act, Vbuf, 256, 512, 2048, 3, S_CACHE_ + SQ_VLM_, SQ_ACT_, NKV_);
    gemm_bf16<float, unsigned short><<<dim3(4, 32), blk, 0, stream>>>(h_act, Wq_act, Qbuf, 256, 2048, 2048, 2, 0, SQ_ACT_, NH_);

    rope_kernel<<<(B_ * NKV_ * SQ_VLM_ * 64 + 255) / 256, 256, 0, stream>>>(
        Kbuf, pos_vlm, NKV_, S_CACHE_, SQ_VLM_, S_TOTAL_, B_ * NKV_ * SQ_VLM_ * 64);
    rope_kernel<<<(B_ * NKV_ * SQ_ACT_ * 64 + 255) / 256, 256, 0, stream>>>(
        Kbuf, pos_act, NKV_, S_CACHE_ + SQ_VLM_, SQ_ACT_, S_TOTAL_, B_ * NKV_ * SQ_ACT_ * 64);
    rope_kernel<<<(B_ * NH_ * SQ_ACT_ * 64 + 255) / 256, 256, 0, stream>>>(
        Qbuf, pos_act, NH_, 0, SQ_ACT_, SQ_ACT_, B_ * NH_ * SQ_ACT_ * 64);

    attn_kernel<<<B_ * NH_, 128, 0, stream>>>(Kbuf, Vbuf, Qbuf, mask, Obuf);

    gemm_bf16<unsigned short, float><<<dim3(4, 32), blk, 0, stream>>>(
        Obuf, Wo_act, out, 256, 2048, 2048, 0, 0, 0, 0);
}
